// Baseline_64080912056948
// MI455X (gfx1250) — matrix operand order repaired and checked by reading, not compiled, not run on hardware
//
#include <hip/hip_runtime.h>
#include <hip/hip_bf16.h>
#include <math.h>

typedef __attribute__((ext_vector_type(16))) _Float16 v16h;
typedef __attribute__((ext_vector_type(8)))  _Float16 h8;
typedef __attribute__((ext_vector_type(8)))  float    v8f;

#define BB 1024
#define TT 256
#define LL 128
#define HH 100
#define KP 128          // all GEMM K dims zero-padded to 128 (compile-time)

// ---------------------------------------------------------------------------
// WMMA GEMM, all-fp16 inputs, fp32 accumulate. K = 128 compile-time.
//   Xh: M x 128 fp16 (row-major, padded)    Wh: Npad x 128 fp16 (row-major,
//   zero-padded)  ->  Y[m,n] = act( sum_k Xh[m,k]*Wh[n,k] + bias[n] )
// One wave per 16x16 tile; no guards in the inner loop: pure b128 loads+WMMA.
// ---------------------------------------------------------------------------
__global__ __launch_bounds__(32)
void wmma_gemm_f16(const _Float16* __restrict__ Xh,
                   const _Float16* __restrict__ Wh,
                   const float* __restrict__ bias, int Nreal,
                   float* __restrict__ Y, int ldy, int do_relu, int ntiles)
{
    const int tm   = blockIdx.x / ntiles;
    const int tn   = blockIdx.x % ntiles;
    const int lane = threadIdx.x;        // 0..31
    const int half = lane >> 4;          // 0 or 1
    const int l16  = lane & 15;

    const _Float16* xrow = Xh + (size_t)(tm * 16 + l16) * KP;  // A row (M=l16)
    const _Float16* wrow = Wh + (size_t)(tn * 16 + l16) * KP;  // B col (N=l16)

    v8f acc = {};
    #pragma unroll
    for (int kb = 0; kb < KP; kb += 32) {
        // A 16x32 layout: half h -> elems 0..15 = K kb+16h .. kb+16h+15 (contig)
        v16h a = *(const v16h*)(xrow + kb + 16 * half);
        // B 32x16 layout: half h -> elems 0..15 = K kb+16h .. kb+16h+15 (contig)
        v16h b = *(const v16h*)(wrow + kb + 16 * half);
        acc = __builtin_amdgcn_wmma_f32_16x16x32_f16(
                  false, a, false, b, (short)0, acc, false, false);
    }

    const int bcol = tn * 16 + l16;
    const float bs = (bcol < Nreal) ? bias[bcol] : 0.0f;
    #pragma unroll
    for (int v = 0; v < 8; ++v) {
        float val = acc[v] + bs;                 // D: VGPR v -> row v + 8*half
        if (do_relu) val = fmaxf(val, 0.0f);
        Y[(size_t)(tm * 16 + v + 8 * half) * ldy + bcol] = val;
    }
}

// ---------------------------------------------------------------------------
// Pack kernels: fp32 -> fp16 (zero-padded) staging.
// ---------------------------------------------------------------------------
__global__ __launch_bounds__(256)
void pack_x_f16(const float* __restrict__ x, _Float16* __restrict__ xh)
{
    const int idx = blockIdx.x * 256 + threadIdx.x;   // 0 .. BB*KP-1
    const int b = idx >> 7, k = idx & 127;
    xh[idx] = (_Float16)x[b * TT + k];
}

__global__ __launch_bounds__(256)
void pack_w_f16(const float* __restrict__ W, int Nreal, int Kreal,
                _Float16* __restrict__ Wh)            // Npad x 128
{
    const int idx = blockIdx.x * 256 + threadIdx.x;
    const int n = idx >> 7, k = idx & 127;
    float v = (n < Nreal && k < Kreal) ? W[n * Kreal + k] : 0.0f;
    Wh[idx] = (_Float16)v;
}

__global__ __launch_bounds__(256)
void zero_act_pad(_Float16* __restrict__ h1h, _Float16* __restrict__ h2h)
{
    const int idx = blockIdx.x * 256 + threadIdx.x;   // 0 .. BB*(KP-HH)-1
    const int i = idx / (KP - HH);
    const int c = HH + idx % (KP - HH);
    h1h[i * KP + c] = (_Float16)0.0f;
    h2h[i * KP + c] = (_Float16)0.0f;
}

// ---------------------------------------------------------------------------
// BatchNorm (training stats): block per feature; writes normalized fp16
// into the padded activation buffer consumed by the next WMMA GEMM.
// ---------------------------------------------------------------------------
__global__ __launch_bounds__(256)
void bn_train_f16out(const float* __restrict__ H, int ldh,
                     const float* __restrict__ gamma,
                     const float* __restrict__ beta,
                     _Float16* __restrict__ Hh)       // M x 128 padded
{
    const int j = blockIdx.x;
    __shared__ float ssum[256];
    __shared__ float ssq[256];
    float s = 0.0f, q = 0.0f;
    for (int i = threadIdx.x; i < BB; i += 256) {
        float v = H[(size_t)i * ldh + j];
        s += v; q += v * v;
    }
    ssum[threadIdx.x] = s; ssq[threadIdx.x] = q;
    __syncthreads();
    for (int off = 128; off > 0; off >>= 1) {
        if ((int)threadIdx.x < off) {
            ssum[threadIdx.x] += ssum[threadIdx.x + off];
            ssq[threadIdx.x]  += ssq[threadIdx.x + off];
        }
        __syncthreads();
    }
    const float mean  = ssum[0] * (1.0f / BB);
    const float var   = ssq[0] * (1.0f / BB) - mean * mean;
    const float scale = gamma[j] * rsqrtf(var + 1e-5f);
    const float shift = beta[j] - mean * scale;
    for (int i = threadIdx.x; i < BB; i += 256) {
        float v = fmaf(H[(size_t)i * ldh + j], scale, shift);
        Hh[(size_t)i * KP + j] = (_Float16)v;
    }
}

// ---------------------------------------------------------------------------
// z = mu + eps*exp(0.5*logvar); accumulate KL sum; emit fp32 + fp16 z.
// ---------------------------------------------------------------------------
__global__ __launch_bounds__(256)
void zkl_kernel(const float* __restrict__ enc,   // (B, 2L)
                const float* __restrict__ eps,   // (B, L)
                float* __restrict__ z_out,
                float* __restrict__ z_ws,
                _Float16* __restrict__ zh,
                float* __restrict__ kl_acc)
{
    const int idx = blockIdx.x * 256 + threadIdx.x;  // 0 .. B*L-1
    const int b = idx >> 7;
    const int l = idx & 127;
    const float mu = enc[b * (2 * LL) + l];
    const float lv = enc[b * (2 * LL) + LL + l];
    const float st = __expf(0.5f * lv);
    const float zv = fmaf(eps[idx], st, mu);
    z_out[idx] = zv;
    z_ws[idx]  = zv;
    zh[idx]    = (_Float16)zv;
    float term = 1.0f + lv - mu * mu - __expf(lv);

    __shared__ float red[256];
    red[threadIdx.x] = term;
    __syncthreads();
    for (int off = 128; off > 0; off >>= 1) {
        if ((int)threadIdx.x < off) red[threadIdx.x] += red[threadIdx.x + off];
        __syncthreads();
    }
    if (threadIdx.x == 0) atomicAdd(kl_acc, red[0]);
}

__global__ void kl_init(float* acc) { acc[0] = 0.0f; }
__global__ void kl_finalize(const float* __restrict__ acc, float* __restrict__ out)
{
    out[0] = -0.5f * acc[0] / (float)BB;
}

// ---------------------------------------------------------------------------
// Fused GRU elementwise + output projection (never materializes the
// (B,255,384)/(B,255,128) broadcast tensors). Block per batch row; 8 waves;
// wave handles t = wave, wave+8, ...; lane caches its 4 j-slices in regs.
// ---------------------------------------------------------------------------
__global__ __launch_bounds__(256)
void gru_out_kernel(const float* __restrict__ x,    // (B, T)
                    const float* __restrict__ z,    // (B, L)
                    const float* __restrict__ gh,   // (B, 3L)
                    const float* __restrict__ wih,  // (3L)
                    const float* __restrict__ bih,  // (3L)
                    const float* __restrict__ Wf,   // (2, L)
                    const float* __restrict__ bf,   // (2)
                    float* __restrict__ pred)       // (B, 2, 255)
{
    const int b    = blockIdx.x;
    const int lane = threadIdx.x & 31;
    const int wave = threadIdx.x >> 5;  // 0..7

    float ar[4], cr[4], az[4], cz[4], an[4], cn[4];
    float ghr[4], ghz[4], ghn[4], zv[4], wf0[4], wf1[4];
    #pragma unroll
    for (int q = 0; q < 4; ++q) {
        const int j = lane + 32 * q;
        ar[q] = wih[j];            cr[q] = bih[j];
        az[q] = wih[LL + j];       cz[q] = bih[LL + j];
        an[q] = wih[2 * LL + j];   cn[q] = bih[2 * LL + j];
        ghr[q] = gh[b * 3 * LL + j];
        ghz[q] = gh[b * 3 * LL + LL + j];
        ghn[q] = gh[b * 3 * LL + 2 * LL + j];
        zv[q]  = z[b * LL + j];
        wf0[q] = Wf[j];
        wf1[q] = Wf[LL + j];
    }
    const float bf0 = bf[0], bf1 = bf[1];

    for (int t = wave; t < TT - 1; t += 8) {
        const float s = x[b * TT + t];
        float s0 = 0.0f, s1 = 0.0f;
        #pragma unroll
        for (int q = 0; q < 4; ++q) {
            float r  = 1.0f / (1.0f + __expf(-(fmaf(s, ar[q], cr[q]) + ghr[q])));
            float zg = 1.0f / (1.0f + __expf(-(fmaf(s, az[q], cz[q]) + ghz[q])));
            float n  = tanhf(fmaf(s, an[q], cn[q]) + r * ghn[q]);
            float hn = fmaf(zg, zv[q] - n, n);   // (1-zg)*n + zg*z
            float h  = fmaxf(hn, 0.0f);
            s0 = fmaf(h, wf0[q], s0);
            s1 = fmaf(h, wf1[q], s1);
        }
        #pragma unroll
        for (int off = 16; off > 0; off >>= 1) {
            s0 += __shfl_xor(s0, off);
            s1 += __shfl_xor(s1, off);
        }
        if (lane == 0) {
            pred[b * 2 * (TT - 1) + t]            = s0 + bf0;
            pred[b * 2 * (TT - 1) + (TT - 1) + t] = s1 + bf1;
        }
    }
}

// ---------------------------------------------------------------------------
extern "C" void kernel_launch(void* const* d_in, const int* in_sizes, int n_in,
                              void* d_out, int out_size, void* d_ws, size_t ws_size,
                              hipStream_t stream)
{
    (void)in_sizes; (void)n_in; (void)out_size; (void)ws_size;

    const float* x     = (const float*)d_in[0];
    const float* eps   = (const float*)d_in[1];
    const float* W1    = (const float*)d_in[2];
    const float* b1    = (const float*)d_in[3];
    const float* g1    = (const float*)d_in[4];
    const float* beta1 = (const float*)d_in[5];
    const float* W2    = (const float*)d_in[6];
    const float* b2    = (const float*)d_in[7];
    const float* g2    = (const float*)d_in[8];
    const float* beta2 = (const float*)d_in[9];
    const float* W3    = (const float*)d_in[10];
    const float* b3    = (const float*)d_in[11];
    const float* Wih   = (const float*)d_in[12];  // (3L,1) -> flat 3L
    const float* Whh   = (const float*)d_in[13];  // (3L, L)
    const float* bih   = (const float*)d_in[14];
    const float* bhh   = (const float*)d_in[15];
    const float* Wf    = (const float*)d_in[16];  // (2, L)
    const float* bf    = (const float*)d_in[17];

    // ---- workspace layout ----
    // fp32 region (floats)
    float* ws    = (float*)d_ws;
    float* h1    = ws;                       // 1024 x 112
    float* h2    = h1  + BB * 112;           // 1024 x 112
    float* enc   = h2  + BB * 112;           // 1024 x 256
    float* gh    = enc + BB * 2 * LL;        // 1024 x 384
    float* zws   = gh  + BB * 3 * LL;        // 1024 x 128
    float* klacc = zws + BB * LL;            // (128-float slot, aligned)
    // fp16 region (halves), starts 32-byte aligned
    _Float16* fh   = (_Float16*)(klacc + 128);
    _Float16* xh   = fh;                     // 1024 x 128
    _Float16* h1h  = xh  + BB * KP;          // 1024 x 128
    _Float16* h2h  = h1h + BB * KP;          // 1024 x 128
    _Float16* zh   = h2h + BB * KP;          // 1024 x 128
    _Float16* W1h  = zh  + BB * KP;          // 112 x 128
    _Float16* W2h  = W1h + 112 * KP;         // 112 x 128
    _Float16* W3h  = W2h + 112 * KP;         // 256 x 128
    _Float16* Whhh = W3h + 256 * KP;         // 384 x 128

    // ---- output layout: predictions (1024*2*255), kl (1), z (1024*128) ----
    float* out   = (float*)d_out;
    const int P  = BB * 2 * (TT - 1);
    float* pred  = out;
    float* klout = out + P;
    float* zout  = out + P + 1;

    kl_init<<<1, 1, 0, stream>>>(klacc);

    // stage fp16 operands (zero-padded)
    pack_x_f16<<<(BB * KP) / 256, 256, 0, stream>>>(x, xh);
    pack_w_f16<<<(112 * KP) / 256, 256, 0, stream>>>(W1, HH, LL, W1h);
    pack_w_f16<<<(112 * KP) / 256, 256, 0, stream>>>(W2, HH, HH, W2h);
    pack_w_f16<<<(256 * KP) / 256, 256, 0, stream>>>(W3, 2 * LL, HH, W3h);
    pack_w_f16<<<(384 * KP) / 256, 256, 0, stream>>>(Whh, 3 * LL, LL, Whhh);
    zero_act_pad<<<(BB * (KP - HH)) / 256 + 1, 256, 0, stream>>>(h1h, h2h);

    // h1 = relu(xin @ W1.T + b1)
    wmma_gemm_f16<<<(BB / 16) * 7, 32, 0, stream>>>(
        xh, W1h, b1, HH, h1, 112, 1, 7);
    bn_train_f16out<<<HH, 256, 0, stream>>>(h1, 112, g1, beta1, h1h);

    // h2 = relu(h1n @ W2.T + b2)
    wmma_gemm_f16<<<(BB / 16) * 7, 32, 0, stream>>>(
        h1h, W2h, b2, HH, h2, 112, 1, 7);
    bn_train_f16out<<<HH, 256, 0, stream>>>(h2, 112, g2, beta2, h2h);

    // enc = h2n @ W3.T + b3
    wmma_gemm_f16<<<(BB / 16) * 16, 32, 0, stream>>>(
        h2h, W3h, b3, 2 * LL, enc, 2 * LL, 0, 16);

    // z, kl
    zkl_kernel<<<(BB * LL) / 256, 256, 0, stream>>>(enc, eps, zout, zws, zh, klacc);
    kl_finalize<<<1, 1, 0, stream>>>(klacc, klout);

    // gh = z @ Whh.T + bhh
    wmma_gemm_f16<<<(BB / 16) * 24, 32, 0, stream>>>(
        zh, Whhh, bhh, 3 * LL, gh, 3 * LL, 0, 24);

    // fused GRU gates + relu + final projection -> predictions
    gru_out_kernel<<<BB, 256, 0, stream>>>(
        x, zws, gh, Wih, bih, Wf, bf, pred);
}
